// MambaBlock_29094108463258
// MI455X (gfx1250) — matrix operand order repaired and checked by reading, not compiled, not run on hardware
//
#include <hip/hip_runtime.h>

// ---------------- problem constants (match reference) ----------------
constexpr int HID = 1024, DIN = 2048, KC = 4, NS = 16, DR = 64;
constexpr int BB = 2, LL = 1024;
constexpr int M_TOK = BB * LL;          // 2048 tokens
constexpr int PROJW = DR + 2 * NS;      // 96

// ---------------- types for WMMA ----------------
typedef __attribute__((ext_vector_type(16))) __bf16 v16bf;
typedef __attribute__((ext_vector_type(8)))  float  v8f;

__device__ __forceinline__ unsigned short f2bf(float f) {
    union { float f; unsigned int u; } c; c.f = f;
    unsigned int u = c.u;
    unsigned int r = u + 0x7FFFu + ((u >> 16) & 1u);   // round-to-nearest-even
    return (unsigned short)(r >> 16);
}

// ---------------- fp32 -> bf16 convert ----------------
__global__ __launch_bounds__(256) void cvt_f32_bf16_kernel(const float* __restrict__ src,
                                                           unsigned short* __restrict__ dst,
                                                           int n) {
    int i = blockIdx.x * blockDim.x + threadIdx.x;
    if (i < n) dst[i] = f2bf(src[i]);
}

// ---------------- bf16 WMMA GEMM:  C[M,N] = A[M,K] * W[N,K]^T  ----------------
// One wave computes a (16*MT) x (16*NT) tile of C with register double-buffering:
// fragments for K-step k+1 are loaded while the WMMAs for step k execute, so the
// s_wait before each WMMA group tolerates a full K-step of outstanding loads.
// Fragment layouts per CDNA5 ISA 7.12.2 (both operands are K-major rows, so each
// lane's fragment is contiguous 16B/32B chunks -> plain global_load_b128).
template <int MT, int NT>
__global__ __launch_bounds__(256) void gemm_bf16_wmma(const unsigned short* __restrict__ A,
                                                      const unsigned short* __restrict__ W,
                                                      float* __restrict__ C,
                                                      int M, int N, int K) {
    const int lane = threadIdx.x & 31;
    const int gw   = (blockIdx.x * blockDim.x + threadIdx.x) >> 5;
    const int tn   = N / (16 * NT);
    const int total = (M / (16 * MT)) * tn;
    if (gw >= total) return;

    const int tm = gw / tn, tj = gw % tn;
    const int m0 = tm * 16 * MT, n0 = tj * 16 * NT;
    const int half = lane >> 4;          // 0: lanes 0-15, 1: lanes 16-31
    const int mr   = lane & 15;

    union Frag { v16bf v; uint4 q[2]; };
    Frag aF[2][MT], bF[2][NT];
    v8f acc[MT][NT];
    v8f vzero = {};
#pragma unroll
    for (int i = 0; i < MT; ++i)
#pragma unroll
        for (int t = 0; t < NT; ++t) acc[i][t] = vzero;

    const unsigned short* aBase = A + (size_t)(m0 + mr) * K + half * 8;
    const unsigned short* bBase = W + (size_t)(n0 + mr) * K + half * 8;

    auto loadFrags = [&](int buf, int k0) {
#pragma unroll
        for (int i = 0; i < MT; ++i) {
            const unsigned short* ap = aBase + (size_t)(i * 16) * K + k0;
            aF[buf][i].q[0] = *reinterpret_cast<const uint4*>(ap);
            aF[buf][i].q[1] = *reinterpret_cast<const uint4*>(ap + 16);
        }
#pragma unroll
        for (int t = 0; t < NT; ++t) {
            const unsigned short* bp = bBase + (size_t)(t * 16) * K + k0;
            bF[buf][t].q[0] = *reinterpret_cast<const uint4*>(bp);
            bF[buf][t].q[1] = *reinterpret_cast<const uint4*>(bp + 16);
        }
    };
    auto wmmaAll = [&](int buf) {
#pragma unroll
        for (int i = 0; i < MT; ++i)
#pragma unroll
            for (int t = 0; t < NT; ++t)
                acc[i][t] = __builtin_amdgcn_wmma_f32_16x16x32_bf16(
                    false, aF[buf][i].v, false, bF[buf][t].v, (short)0,
                    acc[i][t], false, false);
    };

    // software pipeline over K (K is a multiple of 64 for all call sites)
    loadFrags(0, 0);
    int k0 = 0;
    for (; k0 + 64 < K; k0 += 64) {
        loadFrags(1, k0 + 32);
        wmmaAll(0);
        loadFrags(0, k0 + 64);
        wmmaAll(1);
    }
    loadFrags(1, k0 + 32);
    wmmaAll(0);
    wmmaAll(1);

    // store: C layout - VGPR r, lanes 0-15 -> M = r, lanes 16-31 -> M = 8 + r
#pragma unroll
    for (int i = 0; i < MT; ++i) {
        const int rowBase = m0 + i * 16 + half * 8;
#pragma unroll
        for (int t = 0; t < NT; ++t) {
            float* cp = C + (size_t)rowBase * N + n0 + t * 16 + mr;
#pragma unroll
            for (int r = 0; r < 8; ++r) cp[(size_t)r * N] = acc[i][t][r];
        }
    }
}

// ---------------- causal depthwise conv (K=4) + bias + SiLU ----------------
// reads xs-part (cols [0,DIN)) of xz; writes fp32 xs and bf16 xs
__global__ __launch_bounds__(256) void conv_silu_kernel(const float* __restrict__ xz,
                                                        const float* __restrict__ Wc,
                                                        const float* __restrict__ bc,
                                                        float* __restrict__ xs_f32,
                                                        unsigned short* __restrict__ xs_bf16) {
    int idx = blockIdx.x * blockDim.x + threadIdx.x;
    if (idx >= M_TOK * DIN) return;
    int d = idx % DIN;
    int l = (idx / DIN) % LL;
    int b = idx / (DIN * LL);
    float acc = bc[d];
#pragma unroll
    for (int k = 0; k < KC; ++k) {
        int ls = l + k - (KC - 1);
        if (ls >= 0)
            acc += Wc[d * KC + k] * xz[((size_t)(b * LL + ls)) * (2 * DIN) + d];
    }
    float s = acc / (1.f + expf(-acc));       // SiLU
    xs_f32[idx]  = s;
    xs_bf16[idx] = f2bf(s);
}

// ---------------- delta = softplus(proj[:, :DR] @ W_dt^T + b_dt) ----------------
__global__ __launch_bounds__(256) void delta_kernel(const float* __restrict__ proj,
                                                    const float* __restrict__ W_dt,
                                                    const float* __restrict__ b_dt,
                                                    float* __restrict__ delta) {
    int idx = blockIdx.x * blockDim.x + threadIdx.x;
    if (idx >= M_TOK * DIN) return;
    int d = idx % DIN;
    int m = idx / DIN;
    const float* pr = proj + (size_t)m * PROJW;
    const float* wd = W_dt + (size_t)d * DR;
    float acc = b_dt[d];
#pragma unroll 8
    for (int r = 0; r < DR; ++r) acc += pr[r] * wd[r];
    delta[idx] = (acc > 20.f) ? acc : log1pf(expf(acc));   // softplus
}

// ---------------- fused selective scan + skip + gate, emits y in bf16 ----------------
// wave32: lane = c*16 + n ; each wave handles channels {d0, d0+1}, 16 states each.
// Software-pipelined: token l+1's operands are fetched while token l's recurrence
// (exp + shfl-reduce) executes, so the serial chain is ALU-bound, not latency-bound.
__global__ __launch_bounds__(256) void scan_kernel(const float* __restrict__ delta,
                                                   const float* __restrict__ xs,
                                                   const float* __restrict__ proj,
                                                   const float* __restrict__ xz,
                                                   const float* __restrict__ A_log,
                                                   const float* __restrict__ Dvec,
                                                   unsigned short* __restrict__ y_bf16) {
    const int lane = threadIdx.x & 31;
    const int w = (blockIdx.x * blockDim.x + threadIdx.x) >> 5;
    if (w >= BB * (DIN / 2)) return;
    const int b  = w / (DIN / 2);
    const int d0 = (w % (DIN / 2)) * 2;
    const int n  = lane & 15;
    const int d  = d0 + (lane >> 4);

    const float Aneg = -expf(A_log[d * NS + n]);   // A = -exp(A_log)
    const float Dv   = Dvec[d];
    float u = 0.f;

    size_t tok = (size_t)b * LL;
    float dl = delta[tok * DIN + d];
    float xv = xs[tok * DIN + d];
    float Bm = proj[tok * PROJW + DR + n];
    float Cm = proj[tok * PROJW + DR + NS + n];
    float zv = xz[tok * (2 * DIN) + DIN + d];

    for (int l = 0; l < LL; ++l, ++tok) {
        float dl_n = 0.f, xv_n = 0.f, Bm_n = 0.f, Cm_n = 0.f, zv_n = 0.f;
        if (l + 1 < LL) {                         // prefetch next token
            dl_n = delta[(tok + 1) * DIN + d];
            xv_n = xs[(tok + 1) * DIN + d];
            Bm_n = proj[(tok + 1) * PROJW + DR + n];
            Cm_n = proj[(tok + 1) * PROJW + DR + NS + n];
            zv_n = xz[(tok + 1) * (2 * DIN) + DIN + d];
        }
        u = expf(dl * Aneg) * u + (dl * xv) * Bm;  // state update
        float p = u * Cm;
        p += __shfl_xor(p, 1, 16);                 // reduce over 16 states
        p += __shfl_xor(p, 2, 16);
        p += __shfl_xor(p, 4, 16);
        p += __shfl_xor(p, 8, 16);
        const float g  = zv / (1.f + expf(-zv));   // silu(z)
        const float yv = (p + xv * Dv) * g;
        if (n == 0) y_bf16[tok * DIN + d] = f2bf(yv);
        dl = dl_n; xv = xv_n; Bm = Bm_n; Cm = Cm_n; zv = zv_n;
    }
}

// ---------------- host-side orchestration ----------------
extern "C" void kernel_launch(void* const* d_in, const int* in_sizes, int n_in,
                              void* d_out, int out_size, void* d_ws, size_t ws_size,
                              hipStream_t stream) {
    const float* x      = (const float*)d_in[0];
    const float* W_in   = (const float*)d_in[1];
    const float* W_conv = (const float*)d_in[2];
    const float* b_conv = (const float*)d_in[3];
    const float* W_xprj = (const float*)d_in[4];
    const float* W_dt   = (const float*)d_in[5];
    const float* b_dt   = (const float*)d_in[6];
    const float* W_out  = (const float*)d_in[7];
    const float* A_log  = (const float*)d_in[8];
    const float* Dv     = (const float*)d_in[9];
    float* out = (float*)d_out;

    // workspace carve-up (256B aligned)
    char* base = (char*)d_ws;
    size_t off = 0;
    auto carve = [&](size_t bytes) {
        char* p = base + off;
        off = (off + bytes + 255) & ~(size_t)255;
        return p;
    };
    float* xz       = (float*)carve((size_t)M_TOK * 2 * DIN * 4);
    float* xs_f32   = (float*)carve((size_t)M_TOK * DIN * 4);
    float* proj     = (float*)carve((size_t)M_TOK * PROJW * 4);
    float* delta    = (float*)carve((size_t)M_TOK * DIN * 4);
    unsigned short* x_bf    = (unsigned short*)carve((size_t)M_TOK * HID * 2);
    unsigned short* Win_bf  = (unsigned short*)carve((size_t)2 * DIN * HID * 2);
    unsigned short* Wxp_bf  = (unsigned short*)carve((size_t)PROJW * DIN * 2);
    unsigned short* Wout_bf = (unsigned short*)carve((size_t)HID * DIN * 2);
    unsigned short* xs_bf   = (unsigned short*)carve((size_t)M_TOK * DIN * 2);
    unsigned short* y_bf    = (unsigned short*)carve((size_t)M_TOK * DIN * 2);
    (void)ws_size; (void)n_in; (void)in_sizes; (void)out_size;

    auto cvt = [&](const float* s, unsigned short* d, int n) {
        cvt_f32_bf16_kernel<<<(n + 255) / 256, 256, 0, stream>>>(s, d, n);
    };
    cvt(x,      x_bf,    M_TOK * HID);
    cvt(W_in,   Win_bf,  2 * DIN * HID);
    cvt(W_xprj, Wxp_bf,  PROJW * DIN);
    cvt(W_out,  Wout_bf, HID * DIN);

    auto gemm_grid = [](int M, int N, int MT, int NT) {
        int waves = (M / (16 * MT)) * (N / (16 * NT));
        return (waves * 32 + 255) / 256;
    };

    // 1) xz = x @ W_in^T   (2048 x 1024 x 4096), 32x64 tile per wave
    gemm_bf16_wmma<2, 4><<<gemm_grid(M_TOK, 2 * DIN, 2, 4), 256, 0, stream>>>(
        x_bf, Win_bf, xz, M_TOK, 2 * DIN, HID);

    // 2) causal depthwise conv + SiLU
    conv_silu_kernel<<<(M_TOK * DIN + 255) / 256, 256, 0, stream>>>(
        xz, W_conv, b_conv, xs_f32, xs_bf);

    // 3) proj = xs @ W_xproj^T   (2048 x 2048 x 96), 32x32 tile per wave
    gemm_bf16_wmma<2, 2><<<gemm_grid(M_TOK, PROJW, 2, 2), 256, 0, stream>>>(
        xs_bf, Wxp_bf, proj, M_TOK, PROJW, DIN);

    // 4) delta = softplus(proj[:, :64] @ W_dt^T + b_dt)
    delta_kernel<<<(M_TOK * DIN + 255) / 256, 256, 0, stream>>>(
        proj, W_dt, b_dt, delta);

    // 5) selective scan + skip + gate -> y (bf16)
    scan_kernel<<<(BB * (DIN / 2) * 32 + 255) / 256, 256, 0, stream>>>(
        delta, xs_f32, proj, xz, A_log, Dv, y_bf);

    // 6) out = y @ W_out^T   (2048 x 2048 x 1024), 32x64 tile per wave
    gemm_bf16_wmma<2, 4><<<gemm_grid(M_TOK, HID, 2, 4), 256, 0, stream>>>(
        y_bf, Wout_bf, out, M_TOK, HID, DIN);
}
